// CrossScaleAttention_11450382811412
// MI455X (gfx1250) — compile-verified, matrix operand order checked by reading, not run on hardware
//
#include <hip/hip_runtime.h>

typedef __attribute__((ext_vector_type(16))) _Float16 v16h;
typedef __attribute__((ext_vector_type(8)))  float    v8f;

#define BATCH 4
#define CCH   64
#define CRCH  16
#define HH    96
#define WW    96
#define NPIX  (HH*WW)        // 9216
#define HD    48
#define P2    (HD*HD)        // 2304 patches
#define K1DIM 144            // 16*3*3
#define K1PAD 160            // padded to multiple of 32
#define K2DIM (P2*9)         // 20736 ; k' = t*P2 + p ordering
#define OH    192
#define WAVES 8              // waves per block (256 threads)
#define ROWB  (NPIX*2)       // bytes per att p-row = 18432
#define ZPBYTES (112*256*16) // 448 KB zero page (>= 24*ROWB + 2)

// ---- fragment K-offset for CDNA5 16-bit WMMA operand layout -----------------
// pair pv in [0,8): halves 2pv,2pv+1 hold K = base, base+1 with
// base = 16*(pv>>2) + 2*(pv&3) + 8*half   (half = lane>>4)
__device__ __forceinline__ constexpr int frag_k(int pv, int half) {
    return ((pv >> 2) << 4) + ((pv & 3) << 1) + (half << 3);
}

union Frag { v16h v; unsigned int u[8]; _Float16 h[16]; };

// ---- K0: zero the OOB zero-page ---------------------------------------------
__global__ void k_zero(float4* __restrict__ zp) {
    int idx = blockIdx.x * blockDim.x + threadIdx.x;
    zp[idx] = make_float4(0.f, 0.f, 0.f, 0.f);
}

// ---- K1: fused 1x1 convs + PReLU -> asm(f16), q(f16) ------------------------
__global__ void k_front(const float* __restrict__ x, const float* __restrict__ wa,
                        const float* __restrict__ ba, const float* __restrict__ aa,
                        const float* __restrict__ w1, const float* __restrict__ b1,
                        const float* __restrict__ a1,
                        _Float16* __restrict__ asmF, _Float16* __restrict__ qF) {
    int idx = blockIdx.x * blockDim.x + threadIdx.x;   // b*NPIX + n
    int b = idx / NPIX, n = idx - b * NPIX;
    const float* xb = x + (size_t)b * CCH * NPIX + n;
    float accA[CCH], accQ[CRCH];
    #pragma unroll
    for (int o = 0; o < CCH; ++o)  accA[o] = ba[o];
    #pragma unroll
    for (int o = 0; o < CRCH; ++o) accQ[o] = b1[o];
    for (int c = 0; c < CCH; ++c) {
        float xv = xb[(size_t)c * NPIX];
        #pragma unroll
        for (int o = 0; o < CCH; ++o)  accA[o] += wa[o * CCH + c] * xv;
        #pragma unroll
        for (int o = 0; o < CRCH; ++o) accQ[o] += w1[o * CCH + c] * xv;
    }
    float alA = aa[0], alQ = a1[0];
    #pragma unroll
    for (int o = 0; o < CCH; ++o) {
        float v = accA[o]; v = (v >= 0.f) ? v : alA * v;
        asmF[((size_t)b * CCH + o) * NPIX + n] = (_Float16)v;
    }
    #pragma unroll
    for (int o = 0; o < CRCH; ++o) {
        float v = accQ[o]; v = (v >= 0.f) ? v : alQ * v;
        qF[((size_t)b * CRCH + o) * NPIX + n] = (_Float16)v;
    }
}

// ---- K2: 2x2 avg pool + 1x1 conv + PReLU -> kf(f32, 16x48x48) ---------------
__global__ void k_kf(const float* __restrict__ x, const float* __restrict__ w2,
                     const float* __restrict__ b2, const float* __restrict__ a2,
                     float* __restrict__ kf) {
    int idx = blockIdx.x * blockDim.x + threadIdx.x;   // b*P2 + m
    int b = idx / P2, m = idx - b * P2;
    int yy = m / HD, xx = m - yy * HD;
    const float* xb = x + (size_t)b * CCH * NPIX + (2 * yy) * WW + 2 * xx;
    float acc[CRCH];
    #pragma unroll
    for (int o = 0; o < CRCH; ++o) acc[o] = b2[o];
    for (int c = 0; c < CCH; ++c) {
        const float* p = xb + (size_t)c * NPIX;
        float xd = 0.25f * (p[0] + p[1] + p[WW] + p[WW + 1]);
        #pragma unroll
        for (int o = 0; o < CRCH; ++o) acc[o] += w2[o * CCH + c] * xd;
    }
    float al = a2[0];
    #pragma unroll
    for (int o = 0; o < CRCH; ++o) {
        float v = acc[o]; v = (v >= 0.f) ? v : al * v;
        kf[((size_t)b * CRCH + o) * P2 + m] = v;
    }
}

// ---- K3: patch L2-norm + normalized patch matrix wkT[b][p][K1PAD] (f16) -----
__global__ void k_wk(const float* __restrict__ kf, _Float16* __restrict__ wkT) {
    int idx = blockIdx.x * blockDim.x + threadIdx.x;   // b*P2 + p
    int b = idx / P2, p = idx - b * P2;
    int pyp = p / HD, pxp = p - pyp * HD;
    const float* kfb = kf + (size_t)b * CRCH * P2;
    float ss = 0.f;
    for (int c = 0; c < CRCH; ++c)
        #pragma unroll
        for (int i = 0; i < 3; ++i)
            #pragma unroll
            for (int j = 0; j < 3; ++j) {
                int yy = pyp + i - 1, xx = pxp + j - 1;
                float v = ((unsigned)yy < (unsigned)HD && (unsigned)xx < (unsigned)HD)
                              ? kfb[c * P2 + yy * HD + xx] : 0.f;
                ss += v * v;
            }
    float inv = 1.f / fmaxf(sqrtf(ss), 1e-4f);
    _Float16* row = wkT + ((size_t)b * P2 + p) * K1PAD;
    int k = 0;
    for (int c = 0; c < CRCH; ++c)
        #pragma unroll
        for (int i = 0; i < 3; ++i)
            #pragma unroll
            for (int j = 0; j < 3; ++j) {
                int yy = pyp + i - 1, xx = pxp + j - 1;
                float v = ((unsigned)yy < (unsigned)HD && (unsigned)xx < (unsigned)HD)
                              ? kfb[c * P2 + yy * HD + xx] : 0.f;
                row[k++] = (_Float16)(v * inv);
            }
    for (; k < K1PAD; ++k) row[k] = (_Float16)0.f;
}

// ---- K4: im2col of q, transposed: qcolT[b][n][K1PAD] (f16) ------------------
__global__ void k_qcol(const _Float16* __restrict__ qF, _Float16* __restrict__ qcolT) {
    int idx = blockIdx.x * blockDim.x + threadIdx.x;   // b*NPIX + n
    int b = idx / NPIX, n = idx - b * NPIX;
    int y = n / WW, x = n - y * WW;
    const _Float16* qb = qF + (size_t)b * CRCH * NPIX;
    _Float16* row = qcolT + ((size_t)b * NPIX + n) * K1PAD;
    int k = 0;
    for (int c = 0; c < CRCH; ++c)
        #pragma unroll
        for (int i = 0; i < 3; ++i)
            #pragma unroll
            for (int j = 0; j < 3; ++j) {
                int yy = y + i - 1, xx = x + j - 1;
                row[k++] = ((unsigned)yy < (unsigned)HH && (unsigned)xx < (unsigned)WW)
                               ? qb[c * NPIX + yy * WW + xx] : (_Float16)0.f;
            }
    for (; k < K1PAD; ++k) row[k] = (_Float16)0.f;
}

// ---- K5: build aggregation A-matrices A2[par][o][k'] for one sample ---------
// k' = t*P2 + p (tap-major) so GEMM K-steps stay within one (dy,dx) stripe.
// value = asm[s][o][2*(pyp-dy)+py][2*(pxp-dx)+px]  (0 if out of bounds)
__global__ void k_a2(const _Float16* __restrict__ asmF, _Float16* __restrict__ A2, int s) {
    int idx = blockIdx.x * blockDim.x + threadIdx.x;   // (par*64+o)*K2DIM + k'
    int kq = idx % K2DIM;
    int rest = idx / K2DIM;
    int o = rest & 63, par = rest >> 6;
    int t = kq / P2, p = kq - t * P2;
    int dy = t / 3 - 1, dx = t - (t / 3) * 3 - 1;
    int pyp = p / HD, pxp = p - pyp * HD;
    int py = par >> 1, px = par & 1;
    int row = 2 * (pyp - dy) + py, col = 2 * (pxp - dx) + px;
    _Float16 v = (_Float16)0.f;
    if ((unsigned)row < (unsigned)HH && (unsigned)col < (unsigned)WW)
        v = asmF[((size_t)s * CCH + o) * NPIX + row * WW + col];
    A2[idx] = v;
}

// ---- G1: scores = wkT(2304x160) x qcolT^T(160x9216), f16 WMMA ---------------
__global__ void k_gemm_scores(const _Float16* __restrict__ wkTs,
                              const _Float16* __restrict__ qcolTs,
                              _Float16* __restrict__ scores) {
    const int NTN = NPIX / 16;                         // 576
    int wave = threadIdx.x >> 5, lane = threadIdx.x & 31;
    int tile = blockIdx.x * WAVES + wave;
    int tM = tile / NTN, tN = tile - tM * NTN;
    int m = lane & 15, half = lane >> 4;
    const _Float16* Arow = wkTs  + (size_t)(tM * 16 + m) * K1PAD;
    int colN = tN * 16 + m;
    const _Float16* Brow = qcolTs + (size_t)colN * K1PAD;
    v8f acc = {};
    Frag a, b;
    #pragma unroll
    for (int kb = 0; kb < K1PAD; kb += 32) {
        #pragma unroll
        for (int pv = 0; pv < 8; ++pv) {
            int k = kb + frag_k(pv, half);
            a.u[pv] = *(const unsigned int*)(Arow + k);
            b.u[pv] = *(const unsigned int*)(Brow + k);
        }
        acc = __builtin_amdgcn_wmma_f32_16x16x32_f16(false, a.v, false, b.v,
                                                     (short)0, acc, false, false);
    }
    #pragma unroll
    for (int r = 0; r < 8; ++r) {
        int prow = tM * 16 + r + 8 * half;
        scores[(size_t)prow * NPIX + colN] = (_Float16)acc[r];
    }
}

// ---- K6: softmax over the 2304-patch axis, per pixel column, in place -------
__global__ void k_softmax(_Float16* __restrict__ sc) {
    int wave = threadIdx.x >> 5, lane = threadIdx.x & 31;
    int col = blockIdx.x * WAVES + wave;
    float mx = -3.0e38f;
    for (int p = lane; p < P2; p += 32)
        mx = fmaxf(mx, (float)sc[(size_t)p * NPIX + col]);
    #pragma unroll
    for (int o = 16; o; o >>= 1) mx = fmaxf(mx, __shfl_xor(mx, o, 32));
    float sum = 0.f;
    for (int p = lane; p < P2; p += 32)
        sum += __expf(10.f * ((float)sc[(size_t)p * NPIX + col] - mx));
    #pragma unroll
    for (int o = 16; o; o >>= 1) sum += __shfl_xor(sum, o, 32);
    float inv = 1.f / sum;
    for (int p = lane; p < P2; p += 32) {
        float e = __expf(10.f * ((float)sc[(size_t)p * NPIX + col] - mx));
        sc[(size_t)p * NPIX + col] = (_Float16)(e * inv);
    }
}

// ---- G2: aggregation GEMM per parity: out = A2(64x20736) x im2col(att) ------
// Tap-major K: per t-stripe the (dy,dx) shift is fixed, so each lane's B
// addresses are one base pointer + compile-time immediate offsets (p-offset *
// 18432 bytes). OOB lanes point at a zero page with step-advance 0.
__global__ void k_gemm_agg(const _Float16* __restrict__ A2,
                           const _Float16* __restrict__ att,
                           const _Float16* __restrict__ zpage,
                           float* __restrict__ out, int s) {
    const int NTN = NPIX / 16;                         // 576
    int wave = threadIdx.x >> 5, lane = threadIdx.x & 31;
    int tile = blockIdx.x * WAVES + wave;              // par*4*576 + mt*576 + nt
    int nt = tile % NTN;
    int rest = tile / NTN;
    int mt = rest & 3, par = rest >> 2;
    int m = lane & 15, half = lane >> 4;
    int colN = nt * 16 + m;
    int yP = colN / WW, xP = colN - yP * WW;
    int py = par >> 1, px = par & 1;
    const _Float16* Arow = A2 + ((size_t)par * CCH + mt * 16 + m) * K2DIM;
    v8f acc = {};
    Frag a, b;
    for (int t = 0; t < 9; ++t) {
        int dy = t / 3 - 1, dx = t - (t / 3) * 3 - 1;  // wave-uniform
        int yy = yP + dy, xx = xP + dx;                // per-lane
        bool inb = ((unsigned)yy < (unsigned)HH) && ((unsigned)xx < (unsigned)WW);
        const char* bbase = inb ? (const char*)(att + yy * WW + xx) : (const char*)zpage;
        long long stepAdv = inb ? (long long)(32 * ROWB) : 0ll;
        const _Float16* Abase = Arow + t * P2;
        #pragma unroll 2
        for (int step = 0; step < P2 / 32; ++step) {   // 72 K-steps per stripe
            #pragma unroll
            for (int pv = 0; pv < 8; ++pv) {
                const int koff = frag_k(pv, half & 1); // half is per-lane; see below
            }
            // (loads below; koff immediates depend on per-lane `half`, so
            //  split the two halves explicitly to keep offsets compile-time)
            if (half == 0) {
                #pragma unroll
                for (int pv = 0; pv < 8; ++pv) {
                    const int koff = frag_k(pv, 0);
                    a.u[pv] = *(const unsigned int*)(Abase + koff);
                    b.h[2 * pv]     = *(const _Float16*)(bbase + (size_t)koff * ROWB);
                    b.h[2 * pv + 1] = *(const _Float16*)(bbase + (size_t)(koff + 1) * ROWB);
                }
            } else {
                #pragma unroll
                for (int pv = 0; pv < 8; ++pv) {
                    const int koff = frag_k(pv, 1);
                    a.u[pv] = *(const unsigned int*)(Abase + koff);
                    b.h[2 * pv]     = *(const _Float16*)(bbase + (size_t)koff * ROWB);
                    b.h[2 * pv + 1] = *(const _Float16*)(bbase + (size_t)(koff + 1) * ROWB);
                }
            }
            acc = __builtin_amdgcn_wmma_f32_16x16x32_f16(false, a.v, false, b.v,
                                                         (short)0, acc, false, false);
            Abase += 32;
            bbase += stepAdv;
        }
    }
    int Y = 2 * yP + py, X = 2 * xP + px;
    #pragma unroll
    for (int r = 0; r < 8; ++r) {
        int o = mt * 16 + r + 8 * half;
        out[(((size_t)s * CCH + o) * OH + Y) * OH + X] = acc[r] * (1.0f / 6.0f);
    }
}

extern "C" void kernel_launch(void* const* d_in, const int* in_sizes, int n_in,
                              void* d_out, int out_size, void* d_ws, size_t ws_size,
                              hipStream_t stream) {
    const float* x  = (const float*)d_in[0];
    const float* wa = (const float*)d_in[1];
    const float* ba = (const float*)d_in[2];
    const float* aa = (const float*)d_in[3];
    const float* w1 = (const float*)d_in[4];
    const float* b1 = (const float*)d_in[5];
    const float* a1 = (const float*)d_in[6];
    const float* w2 = (const float*)d_in[7];
    const float* b2 = (const float*)d_in[8];
    const float* a2 = (const float*)d_in[9];
    float* out = (float*)d_out;

    char* ws = (char*)d_ws;
    size_t off = 0;
    auto carve = [&](size_t bytes) {
        char* p = ws + off;
        off += (bytes + 255) & ~(size_t)255;
        return p;
    };
    _Float16* asmF  = (_Float16*)carve((size_t)BATCH * CCH * NPIX * 2);       // 4.7 MB
    _Float16* qF    = (_Float16*)carve((size_t)BATCH * CRCH * NPIX * 2);      // 1.2 MB
    float*    kf    = (float*)   carve((size_t)BATCH * CRCH * P2 * 4);        // 0.6 MB
    _Float16* wkT   = (_Float16*)carve((size_t)BATCH * P2 * K1PAD * 2);       // 2.9 MB
    _Float16* qcolT = (_Float16*)carve((size_t)BATCH * NPIX * K1PAD * 2);     // 11.8 MB
    _Float16* A2    = (_Float16*)carve((size_t)4 * CCH * K2DIM * 2);          // 10.6 MB (per-sample reuse)
    _Float16* att   = (_Float16*)carve((size_t)P2 * NPIX * 2);                // 42.5 MB (per-sample reuse)
    _Float16* zp    = (_Float16*)carve((size_t)ZPBYTES);                      // 0.45 MB zero page

    // zero page for OOB B-lanes (ws is not re-zeroed by the harness)
    k_zero <<<112, 256, 0, stream>>>((float4*)zp);

    // front end (all samples)
    k_front<<<(BATCH * NPIX) / 256, 256, 0, stream>>>(x, wa, ba, aa, w1, b1, a1, asmF, qF);
    k_kf   <<<(BATCH * P2)   / 256, 256, 0, stream>>>(x, w2, b2, a2, kf);
    k_wk   <<<(BATCH * P2)   / 256, 256, 0, stream>>>(kf, wkT);
    k_qcol <<<(BATCH * NPIX) / 256, 256, 0, stream>>>(qF, qcolT);

    const int g1_blocks = (P2 / 16) * (NPIX / 16) / WAVES;      // 10368
    const int g2_blocks = 4 * (CCH / 16) * (NPIX / 16) / WAVES; // 1152
    const int sm_blocks = NPIX / WAVES;                         // 1152
    const int a2_blocks = (4 * CCH * K2DIM) / 256;              // 20736

    for (int s = 0; s < BATCH; ++s) {
        k_a2<<<a2_blocks, 256, 0, stream>>>(asmF, A2, s);
        k_gemm_scores<<<g1_blocks, 256, 0, stream>>>(
            wkT + (size_t)s * P2 * K1PAD, qcolT + (size_t)s * NPIX * K1PAD, att);
        k_softmax<<<sm_blocks, 256, 0, stream>>>(att);
        k_gemm_agg<<<g2_blocks, 256, 0, stream>>>(A2, att, zp, out, s);
    }
}
